// Net_49383533969725
// MI455X (gfx1250) — matrix operand order repaired and checked by reading, not compiled, not run on hardware
//
#include <hip/hip_runtime.h>
#include <hip/hip_bf16.h>

typedef __attribute__((ext_vector_type(16))) _Float16 v16h;
typedef __attribute__((ext_vector_type(8)))  float    v8f;
typedef __attribute__((ext_vector_type(4)))  float    v4f;

constexpr int kN   = 100000;
constexpr int kE   = 400000;
constexpr int kDin = 32;
constexpr int kF   = 128;
constexpr int kNP  = 8;
constexpr int kNA  = 50000;

// ---- workspace layout (bytes) ----
constexpr size_t OFF_DINV = 0;                          // kN floats
constexpr size_t OFF_XA   = 400128;                     // kNA*32 floats (32B aligned)
constexpr size_t OFF_PK   = OFF_XA + (size_t)kNA * 32 * 4;  // f16 packed weights

// ---- packed f16 offsets (in halves), base = OFF_PK ----
// W gates: per gate 4096 hi + 4096 lo  (1 kchunk x 8 ctiles x 512)
constexpr int PKH_WSTRIDE = 8192;
// L gates: per gate 32768 hi + 32768 lo (8 kchunks x 8 ctiles x 512)
constexpr int PKH_L       = 24576;
constexpr int PKH_LSTRIDE = 65536;
// W_out: 2048 hi + 2048 lo (4 kchunks x 1 ctile x 512, cols 8..15 zero-padded)
constexpr int PKH_WO      = PKH_L + 3 * PKH_LSTRIDE;    // 221184

constexpr int TS = 132;   // LDS tile row stride (floats): keeps 16B alignment, skews banks

// ================= small prep kernels =================
__global__ void k_deg_init(float* deg) {
  int i = blockIdx.x * 256 + threadIdx.x;
  if (i < kN) deg[i] = 1.0f;                 // self loop
}
__global__ void k_deg_count(const int* __restrict__ dst, float* deg) {
  int e = blockIdx.x * 256 + threadIdx.x;
  if (e < kE) atomicAdd(&deg[dst[e]], 1.0f);
}
__global__ void k_dinv(float* deg) {
  int i = blockIdx.x * 256 + threadIdx.x;
  if (i < kN) deg[i] = rsqrtf(deg[i]);       // in place -> dinv
}
// xa[n] = x[n] * dinv[n]^2  (self-loop term), only rows < kNA are ever read
__global__ void k_agg_init(const float* __restrict__ x, const float* __restrict__ dinv,
                           float* __restrict__ xa) {
  int t = blockIdx.x * 256 + threadIdx.x;    // t < kNA*32
  if (t >= kNA * 32) return;
  int n = t >> 5;
  float dv = dinv[n];
  xa[t] = x[t] * dv * dv;
}
// xa[dst] += x[src] * dinv[src]*dinv[dst]; one wave handles one edge (32 dims)
__global__ void k_agg_edge(const float* __restrict__ x, const int* __restrict__ src,
                           const int* __restrict__ dst, const float* __restrict__ dinv,
                           float* __restrict__ xa) {
  long long t = (long long)blockIdx.x * 256 + threadIdx.x;
  if (t >= (long long)kE * 32) return;
  int e = (int)(t >> 5), d = (int)(t & 31);
  int s = src[e], q = dst[e];
  if (q >= kNA) return;                      // dead destination rows
  float nm = dinv[s] * dinv[q];
  atomicAdd(&xa[q * 32 + d], x[s * 32 + d] * nm);
}

// Pack a row-major KxNsrc f32 matrix into WMMA-B lane layout, f16 hi/lo.
// subtile s = kchunk*ctiles + ctile; dst[s*512 + lane*16 + e]
// lane: N = ctile*16 + (lane&15); element e: K = kchunk*32 + (e>>3)*16 + (lane>>4)*8 + (e&7)
__global__ void k_pack(const float* __restrict__ src, _Float16* __restrict__ hi,
                       _Float16* __restrict__ lo, int nsrc, int ctiles, int total) {
  int t = blockIdx.x * 256 + threadIdx.x;
  if (t >= total) return;
  int s = t >> 9, r = t & 511;
  int lane = r >> 4, e = r & 15;
  int kc = s / ctiles, ct = s % ctiles;
  int K = kc * 32 + (e >> 3) * 16 + (lane >> 4) * 8 + (e & 7);
  int col = ct * 16 + (lane & 15);
  float v = (col < nsrc) ? src[K * nsrc + col] : 0.0f;
  _Float16 h = (_Float16)v;
  hi[t] = h;
  lo[t] = (_Float16)(v - (float)h);
}

// ================= WMMA helpers =================
__device__ __forceinline__ _Float16 hi16(float f) { return (_Float16)f; }
__device__ __forceinline__ _Float16 lo16(float f) {
  _Float16 h = (_Float16)f; return (_Float16)(f - (float)h);
}

// hi/lo split FMA: acc += A*B with ~21-bit mantissa accuracy (3 WMMAs)
__device__ __forceinline__ v8f mm3(v16h ah, v16h al, v16h bh, v16h bl, v8f acc) {
  acc = __builtin_amdgcn_wmma_f32_16x16x32_f16(false, ah, false, bh, (short)0, acc, false, false);
  acc = __builtin_amdgcn_wmma_f32_16x16x32_f16(false, al, false, bh, (short)0, acc, false, false);
  acc = __builtin_amdgcn_wmma_f32_16x16x32_f16(false, ah, false, bl, (short)0, acc, false, false);
  return acc;
}

// Load a 16x32 f32 A-tile (row stride `rs`, starting col `kbase`) into the
// wave32 A-operand layout, converted to f16 hi/lo.
// lane: row = lane&15, half = lane>>4; element e: K = (e>>3)*16 + half*8 + (e&7)
__device__ __forceinline__ void loadA(const float* tile, int rs, int kbase, int lane,
                                      v16h& ah, v16h& al) {
  const int row = lane & 15, kh = lane >> 4;
  const float* p = tile + row * rs + kbase + kh * 8;
  v4f f0 = *(const v4f*)(p);
  v4f f1 = *(const v4f*)(p + 4);
  v4f f2 = *(const v4f*)(p + 16);
  v4f f3 = *(const v4f*)(p + 20);
#pragma unroll
  for (int i = 0; i < 4; ++i) {
    ah[i]      = hi16(f0[i]);  al[i]      = lo16(f0[i]);
    ah[4 + i]  = hi16(f1[i]);  al[4 + i]  = lo16(f1[i]);
    ah[8 + i]  = hi16(f2[i]);  al[8 + i]  = lo16(f2[i]);
    ah[12 + i] = hi16(f3[i]);  al[12 + i] = lo16(f3[i]);
  }
}

__device__ __forceinline__ v16h loadB(const _Float16* p, int lane) {
  return *(const v16h*)(p + lane * 16);
}

// stage-1: c_g tile (16 rows x 16 cols for this wave) = A1 @ Wg + bg -> LDS
__device__ __forceinline__ void gate1(const v16h& a1h, const v16h& a1l,
                                      const _Float16* whi, const _Float16* wlo,
                                      const float* __restrict__ bias,
                                      float* cdst, int w, int lane) {
  const int col = (w << 4) + (lane & 15);
  float b = bias[col];
  v8f acc;
#pragma unroll
  for (int v = 0; v < 8; ++v) acc[v] = b;
  acc = mm3(a1h, a1l, loadB(whi + w * 512, lane), loadB(wlo + w * 512, lane), acc);
  const int rb = (lane >> 4) * 8;
#pragma unroll
  for (int v = 0; v < 8; ++v) cdst[(rb + v) * TS + col] = acc[v];
}

// stage-2/3: [c | h2] (16x256) @ L (256x16 slice) + bias, A-tiles from LDS
__device__ __forceinline__ v8f gemm2(const float* ctile, const float* htile,
                                     const _Float16* Lhi, const _Float16* Llo,
                                     const float* __restrict__ bias, int w, int lane) {
  const int col = (w << 4) + (lane & 15);
  float b = bias[col];
  v8f acc;
#pragma unroll
  for (int v = 0; v < 8; ++v) acc[v] = b;
#pragma unroll
  for (int kc = 0; kc < 4; ++kc) {
    v16h ah, al;
    loadA(ctile, TS, kc * 32, lane, ah, al);
    int sub = kc * 8 + w;
    acc = mm3(ah, al, loadB(Lhi + sub * 512, lane), loadB(Llo + sub * 512, lane), acc);
  }
#pragma unroll
  for (int kc = 0; kc < 4; ++kc) {
    v16h ah, al;
    loadA(htile, TS, kc * 32, lane, ah, al);
    int sub = (4 + kc) * 8 + w;
    acc = mm3(ah, al, loadB(Lhi + sub * 512, lane), loadB(Llo + sub * 512, lane), acc);
  }
  return acc;
}

// ================= fused GRU mega-kernel: 16 rows per block =================
__global__ __launch_bounds__(256) void k_mega(
    const float* __restrict__ xa, const float* __restrict__ Hst,
    const _Float16* __restrict__ pk,
    const float* __restrict__ bz, const float* __restrict__ br, const float* __restrict__ bh,
    const float* __restrict__ Lzb, const float* __restrict__ Lrb, const float* __restrict__ Lhb,
    const float* __restrict__ bo, float* __restrict__ out) {
  __shared__ float s_cz[16 * TS], s_cr[16 * TS], s_ch[16 * TS];
  __shared__ float s_H[16 * TS], s_HR[16 * TS], s_A2[16 * TS];

  const int tid = threadIdx.x, w = tid >> 5, lane = tid & 31;
  const int row0 = blockIdx.x * 16;     // row0 < kNA always (grid = kNA/16)

  // cooperative H tile load (16x128)
#pragma unroll
  for (int i = 0; i < 8; ++i) {
    int idx = tid + i * 256;
    int r = idx >> 7, c = idx & 127;
    s_H[r * TS + c] = Hst[(row0 + r) * kF + c];
  }

  // A1 = xa tile (16x32) in A-operand registers, hi/lo f16
  v16h a1h, a1l;
  loadA(xa + (size_t)row0 * 32, 32, 0, lane, a1h, a1l);

  // stage 1: c_z, c_r, c_h = xa @ W* + b*
  gate1(a1h, a1l, pk + 0 * PKH_WSTRIDE, pk + 0 * PKH_WSTRIDE + 4096, bz, s_cz, w, lane);
  gate1(a1h, a1l, pk + 1 * PKH_WSTRIDE, pk + 1 * PKH_WSTRIDE + 4096, br, s_cr, w, lane);
  gate1(a1h, a1l, pk + 2 * PKH_WSTRIDE, pk + 2 * PKH_WSTRIDE + 4096, bh, s_ch, w, lane);
  __syncthreads();

  const int col = (w << 4) + (lane & 15);
  const int rb = (lane >> 4) * 8;

  // stage 2: Z, R
  v8f Z = gemm2(s_cz, s_H, pk + PKH_L + 0 * PKH_LSTRIDE,
                pk + PKH_L + 0 * PKH_LSTRIDE + 32768, Lzb, w, lane);
  v8f R = gemm2(s_cr, s_H, pk + PKH_L + 1 * PKH_LSTRIDE,
                pk + PKH_L + 1 * PKH_LSTRIDE + 32768, Lrb, w, lane);
#pragma unroll
  for (int v = 0; v < 8; ++v) {
    Z[v] = 1.0f / (1.0f + expf(-Z[v]));
    float r = 1.0f / (1.0f + expf(-R[v]));
    s_HR[(rb + v) * TS + col] = s_H[(rb + v) * TS + col] * r;
  }
  __syncthreads();

  // stage 3: H_tilde, H_new, relu
  v8f T = gemm2(s_ch, s_HR, pk + PKH_L + 2 * PKH_LSTRIDE,
                pk + PKH_L + 2 * PKH_LSTRIDE + 32768, Lhb, w, lane);
#pragma unroll
  for (int v = 0; v < 8; ++v) {
    float t = tanhf(T[v]);
    float hv = s_H[(rb + v) * TS + col];
    float hn = Z[v] * hv + (1.0f - Z[v]) * t;
    s_A2[(rb + v) * TS + col] = fmaxf(hn, 0.0f);
  }
  __syncthreads();

  // stage 4: y = relu(H_new) @ W_out + b_out  (16x8, wave 0 only, padded to 16 cols)
  if (w == 0) {
    int oc = lane & 15;
    float b = (oc < kNP) ? bo[oc] : 0.0f;
    v8f acc;
#pragma unroll
    for (int v = 0; v < 8; ++v) acc[v] = b;
#pragma unroll
    for (int kc = 0; kc < 4; ++kc) {
      v16h ah, al;
      loadA(s_A2, TS, kc * 32, lane, ah, al);
      acc = mm3(ah, al, loadB(pk + PKH_WO + kc * 512, lane),
                loadB(pk + PKH_WO + 2048 + kc * 512, lane), acc);
    }
    if (oc < kNP) {
#pragma unroll
      for (int v = 0; v < 8; ++v) out[(size_t)(row0 + rb + v) * kNP + oc] = acc[v];
    }
  }
}

// ================= host launcher =================
extern "C" void kernel_launch(void* const* d_in, const int* in_sizes, int n_in,
                              void* d_out, int out_size, void* d_ws, size_t ws_size,
                              hipStream_t stream) {
  const float* x    = (const float*)d_in[0];
  const float* H    = (const float*)d_in[1];
  const int*   ei   = (const int*)d_in[2];
  const float* Wz   = (const float*)d_in[3];  const float* bz   = (const float*)d_in[4];
  const float* Wr   = (const float*)d_in[5];  const float* br   = (const float*)d_in[6];
  const float* Wh   = (const float*)d_in[7];  const float* bh   = (const float*)d_in[8];
  const float* Lzw  = (const float*)d_in[9];  const float* Lzb  = (const float*)d_in[10];
  const float* Lrw  = (const float*)d_in[11]; const float* Lrb  = (const float*)d_in[12];
  const float* Lhw  = (const float*)d_in[13]; const float* Lhb  = (const float*)d_in[14];
  const float* Wo   = (const float*)d_in[15]; const float* bo   = (const float*)d_in[16];
  float* out = (float*)d_out;

  const int* src = ei;
  const int* dst = ei + kE;

  float*     dinv = (float*)((char*)d_ws + OFF_DINV);
  float*     xa   = (float*)((char*)d_ws + OFF_XA);
  _Float16*  pk   = (_Float16*)((char*)d_ws + OFF_PK);

  // degrees / dinv
  k_deg_init<<<(kN + 255) / 256, 256, 0, stream>>>(dinv);
  k_deg_count<<<(kE + 255) / 256, 256, 0, stream>>>(dst, dinv);
  k_dinv<<<(kN + 255) / 256, 256, 0, stream>>>(dinv);

  // xa = D^-1/2 (A+I) D^-1/2 x   (only dst rows < kNA kept)
  k_agg_init<<<(kNA * 32) / 256, 256, 0, stream>>>(x, dinv, xa);
  k_agg_edge<<<(kE * 32) / 256, 256, 0, stream>>>(x, src, dst, dinv, xa);

  // pack weights into WMMA-B f16 hi/lo layout
  k_pack<<<16, 256, 0, stream>>>(Wz, pk + 0 * PKH_WSTRIDE, pk + 0 * PKH_WSTRIDE + 4096, kF, 8, 4096);
  k_pack<<<16, 256, 0, stream>>>(Wr, pk + 1 * PKH_WSTRIDE, pk + 1 * PKH_WSTRIDE + 4096, kF, 8, 4096);
  k_pack<<<16, 256, 0, stream>>>(Wh, pk + 2 * PKH_WSTRIDE, pk + 2 * PKH_WSTRIDE + 4096, kF, 8, 4096);
  k_pack<<<128, 256, 0, stream>>>(Lzw, pk + PKH_L + 0 * PKH_LSTRIDE,
                                  pk + PKH_L + 0 * PKH_LSTRIDE + 32768, kF, 8, 32768);
  k_pack<<<128, 256, 0, stream>>>(Lrw, pk + PKH_L + 1 * PKH_LSTRIDE,
                                  pk + PKH_L + 1 * PKH_LSTRIDE + 32768, kF, 8, 32768);
  k_pack<<<128, 256, 0, stream>>>(Lhw, pk + PKH_L + 2 * PKH_LSTRIDE,
                                  pk + PKH_L + 2 * PKH_LSTRIDE + 32768, kF, 8, 32768);
  k_pack<<<8, 256, 0, stream>>>(Wo, pk + PKH_WO, pk + PKH_WO + 2048, kNP, 1, 2048);

  // fused GRU + output head over the live 50000 rows
  k_mega<<<kNA / 16, 256, 0, stream>>>(xa, H, pk, bz, br, bh, Lzb, Lrb, Lhb, bo, out);
}
